// SelfAttention_10462540333348
// MI455X (gfx1250) — matrix operand order repaired and checked by reading, not compiled, not run on hardware
//
#include <hip/hip_runtime.h>
#include <hip/hip_bf16.h>

// ---------------------------------------------------------------------------
// Fused batch-softmax self-attention for MI455X (gfx1250, wave32).
//
// Phase 1 (f32 WMMA 16x16x4, f32 in -> f16 out):
//   f_nk[b][n][k] = (f_w @ x + f_b)^T    (f16, k contiguous)  -> s A-operand
//   g_nk[b][n][k] = (g_w @ x + g_b)^T    (f16, k contiguous)  -> s B-operand
//   h_dn[b][d][n] =  h_w @ x + h_b       (f16, n contiguous)  -> o A-operand
//
// Phase 2 (f16 WMMA 16x16x32, f32 accumulate), fused:
//   s[b,p,q] = sum_k f[b,p,k] * g[b,k,q]         one WMMA per 16x16 tile
//   beta     = softmax over b (4 values, register resident)
//   o[b,d,p] = sum_q h[b,d,q] * beta[b,p,q]      K=32 per WMMA
//   out      = lambd * o + x                     (f32 epilogue)
//
// s/beta (256 MB) never touch HBM: each workgroup owns 32 p-rows and streams
// q in 128-wide chunks; beta passes through 34 KB of LDS as f16, where the
// f16 B-operand k order used here (contiguous K-run per lane) makes the "transpose"
// a plain row-segment ds_load_b128.
// ---------------------------------------------------------------------------

typedef float    v2f  __attribute__((ext_vector_type(2)));
typedef float    v8f  __attribute__((ext_vector_type(8)));
typedef _Float16 v8h  __attribute__((ext_vector_type(8)));
typedef _Float16 v16h __attribute__((ext_vector_type(16)));

__device__ __forceinline__ v8f wmma_f32(v2f a, v2f b, v8f c) {
  return __builtin_amdgcn_wmma_f32_16x16x4_f32(false, a, false, b, (short)0, c,
                                               false, false);
}
__device__ __forceinline__ v8f wmma_f16(v16h a, v16h b, v8f c) {
  return __builtin_amdgcn_wmma_f32_16x16x32_f16(false, a, false, b, (short)0, c,
                                                false, false);
}

// ---------------------------------------------------------------------------
// Phase 1: acc = W[m,:] @ x[b,:,n] + bias  (f32 WMMA), store f16.
// transpose==0: out[b][m][n] ; transpose==1: out[b][n][m] (k contiguous)
// ---------------------------------------------------------------------------
__global__ void __launch_bounds__(256)
gemm_wxb(const float* __restrict__ W, const float* __restrict__ bias,
         const float* __restrict__ x, _Float16* __restrict__ out,
         int M, int C, int N, int transpose) {
  const int lane = threadIdx.x & 31;
  const int wave = threadIdx.x >> 5;
  const int g16  = lane >> 4;
  const int l16  = lane & 15;
  const int n0   = (blockIdx.x * 8 + wave) * 16;
  const int m0   = blockIdx.y * 16;
  const int b    = blockIdx.z;

  const float* xb = x + (size_t)b * C * N;
  v8f acc = {};
  for (int c0 = 0; c0 < C; c0 += 4) {
    v2f a = *(const v2f*)(W + (size_t)(m0 + l16) * C + c0 + 2 * g16);
    v2f bb;
    bb.x = xb[(size_t)(c0 + 2 * g16 + 0) * N + n0 + l16];
    bb.y = xb[(size_t)(c0 + 2 * g16 + 1) * N + n0 + l16];
    acc = wmma_f32(a, bb, acc);
  }
  // C/D layout: element (m = r + 8*g16, n = l16) in VGPR r
  if (transpose) {
    _Float16* ob = out + (size_t)b * N * M;
#pragma unroll
    for (int r = 0; r < 8; ++r) {
      int m = m0 + r + 8 * g16;
      ob[(size_t)(n0 + l16) * M + m] = (_Float16)(acc[r] + bias[m]);
    }
  } else {
    _Float16* ob = out + (size_t)b * M * N;
#pragma unroll
    for (int r = 0; r < 8; ++r) {
      int m = m0 + r + 8 * g16;
      ob[(size_t)m * N + n0 + l16] = (_Float16)(acc[r] + bias[m]);
    }
  }
}

// ---------------------------------------------------------------------------
// Phase 2: fused s -> batch-softmax -> o  (f16 WMMA 16x16x32).
// Grid: N/32 blocks; block = 256 threads = 8 waves.
//   producer: wave w builds beta for q-subtile w of the 128-wide chunk,
//             all 4 batches (softmax over b elementwise on accumulators).
//   consumer: wave w owns batch (w>>1), d-tiles (w&1)*8..+7, both p-subtiles.
// ---------------------------------------------------------------------------
#define JCHUNK   128
#define LDS_JS_H 136   // beta row pitch in halves (272 B = 17 * 16 B)

__global__ void __launch_bounds__(256)
attn_fused(const _Float16* __restrict__ f_nk, const _Float16* __restrict__ g_nk,
           const _Float16* __restrict__ h_dn, const float* __restrict__ x,
           const float* __restrict__ lambd_p, float* __restrict__ out,
           int C, int N, int K) {
  __shared__ __align__(16) _Float16 beta_lds[4 * 32 * LDS_JS_H];  // 34,816 B

  const int lane = threadIdx.x & 31;
  const int wave = threadIdx.x >> 5;  // 0..7
  const int g16  = lane >> 4;
  const int l16  = lane & 15;
  const int i0   = blockIdx.x * 32;   // p-block

  const int b_c = wave >> 1;          // consumer batch
  const int dt0 = (wave & 1) * 8;     // consumer d-tile base

  v8f acc[2][8] = {};                 // [p-subtile][d-tile]

  const size_t bNK = (size_t)N * K;
  const size_t bCN = (size_t)C * N;
  const _Float16* hb = h_dn + (size_t)b_c * bCN;

  for (int j0 = 0; j0 < N; j0 += JCHUNK) {
    // ---------------- producer: beta for q-subtile 'wave' ----------------
    const int js = j0 + wave * 16;
#pragma unroll
    for (int it = 0; it < 2; ++it) {
      const int ii = i0 + it * 16;
      v8f s[4];
#pragma unroll
      for (int b = 0; b < 4; ++b) {
        const _Float16* fb = f_nk + (size_t)b * bNK;
        const _Float16* gb = g_nk + (size_t)b * bNK;
        // A (16x32 f16): lane row p = ii+l16; K-run 16*g16..+15
        v16h a, bb;
        const _Float16* fp = fb + (size_t)(ii + l16) * K;
        ((v8h*)&a)[0] = *(const v8h*)(fp + 16 * g16);
        ((v8h*)&a)[1] = *(const v8h*)(fp + 16 * g16 + 8);
        // B (32x16 f16): lane col q = js+l16; K-run 16*g16..+15
        const _Float16* gp = gb + (size_t)(js + l16) * K + 16 * g16;
        ((v8h*)&bb)[0] = *(const v8h*)(gp);
        ((v8h*)&bb)[1] = *(const v8h*)(gp + 8);
        v8f z = {};
        s[b] = wmma_f16(a, bb, z);
      }
      // softmax over the batch axis (4 values per element)
#pragma unroll
      for (int e = 0; e < 8; ++e) {
        float m  = fmaxf(fmaxf(s[0][e], s[1][e]), fmaxf(s[2][e], s[3][e]));
        float e0 = __expf(s[0][e] - m), e1 = __expf(s[1][e] - m);
        float e2 = __expf(s[2][e] - m), e3 = __expf(s[3][e] - m);
        float inv = 1.0f / (e0 + e1 + e2 + e3);
        s[0][e] = e0 * inv; s[1][e] = e1 * inv;
        s[2][e] = e2 * inv; s[3][e] = e3 * inv;
      }
      // beta_lds[b][it*16 + p_local][wave*16 + q_local] as f16
#pragma unroll
      for (int b = 0; b < 4; ++b) {
        _Float16* base =
            beta_lds + (size_t)(b * 32 + it * 16) * LDS_JS_H + wave * 16;
#pragma unroll
        for (int r = 0; r < 8; ++r)
          base[(r + 8 * g16) * LDS_JS_H + l16] = (_Float16)s[b][r];
      }
    }
    __syncthreads();

    // ---------------- consumer: o += h @ beta^T (K=32 per WMMA) ----------
#pragma unroll
    for (int jB = 0; jB < JCHUNK; jB += 32) {
      // B tiles: col = p = l16 (per p-subtile), K-run = jB + 16*g16 .. +15
      v16h B0, B1;
      const _Float16* bl0 =
          beta_lds + (size_t)(b_c * 32 + l16) * LDS_JS_H + jB + 16 * g16;
      const _Float16* bl1 = bl0 + 16 * LDS_JS_H;
      ((v8h*)&B0)[0] = *(const v8h*)(bl0);
      ((v8h*)&B0)[1] = *(const v8h*)(bl0 + 8);
      ((v8h*)&B1)[0] = *(const v8h*)(bl1);
      ((v8h*)&B1)[1] = *(const v8h*)(bl1 + 8);
#pragma unroll
      for (int t = 0; t < 8; ++t) {
        const _Float16* hp =
            hb + (size_t)((dt0 + t) * 16 + l16) * N + j0 + jB;
        v16h a;
        ((v8h*)&a)[0] = *(const v8h*)(hp + 16 * g16);
        ((v8h*)&a)[1] = *(const v8h*)(hp + 16 * g16 + 8);
        acc[0][t] = wmma_f16(a, B0, acc[0][t]);
        acc[1][t] = wmma_f16(a, B1, acc[1][t]);
      }
    }
    __syncthreads();
  }

  // ---------------- epilogue: out = lambd*o + x ---------------------------
  const float lam = lambd_p[0];
  const float* xb = x + (size_t)b_c * bCN;
  float*       ob = out + (size_t)b_c * bCN;
#pragma unroll
  for (int it = 0; it < 2; ++it) {
    const int icol = i0 + it * 16 + l16;
#pragma unroll
    for (int t = 0; t < 8; ++t) {
#pragma unroll
      for (int r = 0; r < 8; ++r) {
        const int d = (dt0 + t) * 16 + r + 8 * g16;
        ob[(size_t)d * N + icol] = lam * acc[it][t][r] + xb[(size_t)d * N + icol];
      }
    }
  }
}

// ---------------------------------------------------------------------------
extern "C" void kernel_launch(void* const* d_in, const int* in_sizes, int n_in,
                              void* d_out, int out_size, void* d_ws, size_t ws_size,
                              hipStream_t stream) {
  (void)in_sizes; (void)n_in; (void)out_size; (void)ws_size;
  const float* x     = (const float*)d_in[0];
  const float* f_w   = (const float*)d_in[1];
  const float* f_b   = (const float*)d_in[2];
  const float* g_w   = (const float*)d_in[3];
  const float* g_b   = (const float*)d_in[4];
  const float* h_w   = (const float*)d_in[5];
  const float* h_b   = (const float*)d_in[6];
  const float* lambd = (const float*)d_in[7];
  float* out = (float*)d_out;

  const int B = 4, C = 256, N = 4096, K = 32;

  // f16 workspace: f_nk (1 MB) | g_nk (1 MB) | h_dn (8 MB)
  _Float16* f_nk = (_Float16*)d_ws;
  _Float16* g_nk = f_nk + (size_t)B * N * K;
  _Float16* h_dn = g_nk + (size_t)B * N * K;

  dim3 blk(256);
  gemm_wxb<<<dim3(N / 128, K / 16, B), blk, 0, stream>>>(f_w, f_b, x, f_nk, K, C, N, 1);
  gemm_wxb<<<dim3(N / 128, K / 16, B), blk, 0, stream>>>(g_w, g_b, x, g_nk, K, C, N, 1);
  gemm_wxb<<<dim3(N / 128, C / 16, B), blk, 0, stream>>>(h_w, h_b, x, h_dn, C, C, N, 0);
  attn_fused<<<dim3(N / 32), blk, 0, stream>>>(f_nk, g_nk, h_dn, x, lambd, out, C, N, K);
}
